// SelfAttention_52948356825334
// MI455X (gfx1250) — matrix operand order repaired and checked by reading, not compiled, not run on hardware
//
#include <hip/hip_runtime.h>
#include <hip/hip_bf16.h>

#define N_SEQ  2048
#define C_DIM  768
#define N_HEAD 16
#define D_HEAD 48
#define D_PAD  64

typedef __attribute__((ext_vector_type(16))) __bf16 bf16x16;
typedef __attribute__((ext_vector_type(8)))  __bf16 bf16x8;
typedef __attribute__((ext_vector_type(4)))  __bf16 bf16x4;
typedef __attribute__((ext_vector_type(8)))  float  f32x8;
typedef __attribute__((ext_vector_type(4)))  float  f32x4;

static __device__ __forceinline__ f32x8 wmma_bf16(bf16x16 a, bf16x16 b, f32x8 c) {
  // D = A(16x32 bf16) * B(32x16 bf16) + C(16x16 f32)
  return __builtin_amdgcn_wmma_f32_16x16x32_bf16(false, a, false, b, (short)0, c,
                                                 false, false);
}

static __device__ __forceinline__ void async_b128(unsigned lds_dst,
                                                  const void* gsrc) {
  asm volatile("global_load_async_to_lds_b128 %0, %1, off"
               :: "v"(lds_dst), "v"((unsigned long long)(size_t)gsrc)
               : "memory");
}

// ---------------------------------------------------------------------------
// Zero padded Q/K/Vt staging (d padded 48 -> 64; pads contribute 0 to Q.K^T).
// ---------------------------------------------------------------------------
__global__ void zero_ws(uint4* __restrict__ p, size_t n16) {
  size_t i = (size_t)blockIdx.x * blockDim.x + threadIdx.x;
  if (i < n16) p[i] = make_uint4(0u, 0u, 0u, 0u);
}

// fp32 -> bf16 elementwise (sizes are multiples of 4)
__global__ void cvt_bf16(const float* __restrict__ src, __bf16* __restrict__ dst,
                         int n4) {
  int i = blockIdx.x * blockDim.x + threadIdx.x;
  if (i >= n4) return;
  f32x4 v = ((const f32x4*)src)[i];
  bf16x4 o;
  #pragma unroll
  for (int e = 0; e < 4; ++e) o[e] = (__bf16)v[e];
  ((bf16x4*)dst)[i] = o;
}

// ---------------------------------------------------------------------------
// Projection GEMMs: Y = X @ W^T (+b).  blockIdx.z: 0=Q 1=K 2=V 3=Gate.
// Double-buffered async-LDS pipeline: X tile (128x32 bf16) + shared W tile
// (64x32 bf16) streamed by the async copy engine; inner loop is pure
// ds_load + WMMA with the global stream in the background.
// ---------------------------------------------------------------------------
__global__ __launch_bounds__(256) void proj_kernel(
    const __bf16* __restrict__ xb1, const __bf16* __restrict__ xb2,
    const __bf16* __restrict__ wb,  // [4][768][768] bf16 (Q,K,V,G)
    const float*  __restrict__ bq,
    __bf16* __restrict__ qbuf,   // [h][n][D_PAD]
    __bf16* __restrict__ kbuf,   // [h][n][D_PAD]
    __bf16* __restrict__ vtbuf,  // [h][D_PAD][n]
    float*  __restrict__ gate)   // [n][c]
{
  __shared__ __bf16 ldsX[2][128][32];  // 2 x 8 KB
  __shared__ __bf16 ldsW[2][64][32];   // 2 x 4 KB
  const int lane = threadIdx.x & 31;
  const int wave = threadIdx.x >> 5;
  const int hi   = lane >> 4;
  const int lo   = lane & 15;
  const int proj = blockIdx.z;
  const int i0b  = blockIdx.x * 128;               // block row tile
  const int j0   = blockIdx.y * 64;                // output col tile

  const __bf16* X = (proj == 0 || proj == 3) ? xb1 : xb2;
  const __bf16* W = wb + (size_t)proj * C_DIM * C_DIM;

  // async tile issue: each lane moves 16B; X: 2 instrs/wave, W: 1 instr/wave
  const int lrow = lane >> 2;          // 0..7
  const int lcol = (lane & 3) * 8;     // bf16 elems, 16B chunks
  auto issue_tiles = [&](int kk, int buf) {
    #pragma unroll
    for (int g = 0; g < 2; ++g) {
      const int rl = wave * 16 + g * 8 + lrow;     // 0..127
      async_b128((unsigned)(size_t)&ldsX[buf][rl][lcol],
                 X + (size_t)(i0b + rl) * C_DIM + kk + lcol);
    }
    const int rw = wave * 8 + lrow;                // 0..63
    async_b128((unsigned)(size_t)&ldsW[buf][rw][lcol],
               W + (size_t)(j0 + rw) * C_DIM + kk + lcol);
  };

  f32x8 acc[4] = {};
  issue_tiles(0, 0);

  for (int kk = 0, it = 0; kk < C_DIM; kk += 32, ++it) {
    const int buf = it & 1;
    asm volatile("s_wait_asynccnt 0x0" ::: "memory");  // own tile slices done
    __syncthreads();                                    // everyone's done
    if (kk + 32 < C_DIM) issue_tiles(kk + 32, buf ^ 1); // safe after barrier

    bf16x16 a;  // A[M=lo][K = kk + (e&7) + ((e>>3)<<4) + (hi<<3)]
    {
      const __bf16* ax = &ldsX[buf][wave * 16 + lo][hi << 3];
      bf16x8 a0 = *(const bf16x8*)ax;
      bf16x8 a1 = *(const bf16x8*)(ax + 16);
      #pragma unroll
      for (int e = 0; e < 8; ++e) { a[e] = a0[e]; a[8 + e] = a1[e]; }
    }
    #pragma unroll
    for (int j = 0; j < 4; ++j) {
      // B[K = kk + (e&7) + ((e>>3)<<4) + (hi<<3)][N = j0+16j+lo]
      bf16x16 bfr;
      {
        const __bf16* bw = &ldsW[buf][16 * j + lo][hi << 3];
        bf16x8 b0 = *(const bf16x8*)bw;
        bf16x8 b1 = *(const bf16x8*)(bw + 16);
        #pragma unroll
        for (int e = 0; e < 8; ++e) { bfr[e] = b0[e]; bfr[8 + e] = b1[e]; }
      }
      acc[j] = wmma_bf16(a, bfr, acc[j]);
    }
  }

  const float qscale = 0.14433756729740643f;  // 1/sqrt(48)
  #pragma unroll
  for (int j = 0; j < 4; ++j) {
    #pragma unroll
    for (int r = 0; r < 8; ++r) {
      const int row = i0b + wave * 16 + r + (hi << 3);  // C: M = r + 8*hi
      const int col = j0 + 16 * j + lo;                 //    N = lo
      float y = acc[j][r];
      if (proj == 0) {
        y = (y + bq[col]) * qscale;
        const int head = col / D_HEAD, dd = col % D_HEAD;
        qbuf[(size_t)(head * N_SEQ + row) * D_PAD + dd] = (__bf16)y;
      } else if (proj == 1) {
        const int head = col / D_HEAD, dd = col % D_HEAD;
        kbuf[(size_t)(head * N_SEQ + row) * D_PAD + dd] = (__bf16)y;
      } else if (proj == 2) {
        const int head = col / D_HEAD, dd = col % D_HEAD;
        vtbuf[(size_t)(head * D_PAD + dd) * N_SEQ + row] = (__bf16)y;
      } else {
        gate[(size_t)row * C_DIM + col] = 1.0f / (1.0f + __expf(-y));
      }
    }
  }
}

// ---------------------------------------------------------------------------
// Flash attention; pair_logits (268 MB, the roofline term) streamed by the
// async copy engine, double-buffered per wave, issued one k-slab ahead.
// ---------------------------------------------------------------------------
__global__ __launch_bounds__(256) void attn_kernel(
    const __bf16* __restrict__ qbuf, const __bf16* __restrict__ kbuf,
    const __bf16* __restrict__ vtbuf, const float* __restrict__ gate,
    const float* __restrict__ pair, const unsigned char* __restrict__ mask,
    float* __restrict__ out)
{
  __shared__ float lds_pair[8][2][16][32];  // per-wave double-buffered tile
  __shared__ float lds_p[8][16][33];        // per-wave P re-layout staging
  const int lane = threadIdx.x & 31;
  const int wave = threadIdx.x >> 5;
  const int hi = lane >> 4, lo = lane & 15;
  const int head = blockIdx.y;
  const int q0   = blockIdx.x * 128 + wave * 16;

  const size_t pair_base = (size_t)head * N_SEQ * N_SEQ;
  const int prow = (lane >> 3);        // + 4*g
  const int pcol = (lane & 7) * 4;
  auto issue_pair = [&](int k0, int buf) {
    #pragma unroll
    for (int g = 0; g < 4; ++g) {
      const int row = g * 4 + prow;
      async_b128((unsigned)(size_t)&lds_pair[wave][buf][row][pcol],
                 pair + pair_base + (size_t)(q0 + row) * N_SEQ + k0 + pcol);
    }
  };

  // Q A-fragments for d = [0,32) and [32,64)
  bf16x16 qa[2];
  {
    const __bf16* qp = qbuf + (size_t)(head * N_SEQ + q0 + lo) * D_PAD;
    #pragma unroll
    for (int w = 0; w < 2; ++w) {
      const bf16x8* p = (const bf16x8*)(qp + 32 * w + (hi << 3));
      bf16x8 lo8 = p[0], hi8 = p[2];  // +16 elements
      #pragma unroll
      for (int e = 0; e < 8; ++e) { qa[w][e] = lo8[e]; qa[w][8 + e] = hi8[e]; }
    }
  }

  float m_r[8], l_r[8];
  #pragma unroll
  for (int r = 0; r < 8; ++r) { m_r[r] = -3.0e38f; l_r[r] = 0.0f; }
  f32x8 o[3] = {};

  issue_pair(0, 0);  // prologue

  for (int k0 = 0; k0 < N_SEQ; k0 += 32) {
    const int buf = (k0 >> 5) & 1;
    if (k0 + 32 < N_SEQ) {
      issue_pair(k0 + 32, buf ^ 1);
      asm volatile("s_wait_asynccnt 0x4" ::: "memory");  // current buf ready
    } else {
      asm volatile("s_wait_asynccnt 0x0" ::: "memory");
    }

    // S = Q K^T for two 16-key subtiles
    f32x8 s[2] = {};
    #pragma unroll
    for (int nt = 0; nt < 2; ++nt) {
      const __bf16* kp =
          kbuf + (size_t)(head * N_SEQ + k0 + 16 * nt + lo) * D_PAD + (hi << 3);
      #pragma unroll
      for (int w = 0; w < 2; ++w) {
        bf16x16 kb;
        {
          bf16x8 k0v = *(const bf16x8*)(kp + 32 * w);
          bf16x8 k1v = *(const bf16x8*)(kp + 32 * w + 16);
          #pragma unroll
          for (int e = 0; e < 8; ++e) { kb[e] = k0v[e]; kb[8 + e] = k1v[e]; }
        }
        s[nt] = wmma_bf16(qa[w], kb, s[nt]);
      }
    }

    // + pair bias (from LDS tile), key mask
    #pragma unroll
    for (int nt = 0; nt < 2; ++nt) {
      const int kcol = k0 + 16 * nt + lo;
      const bool mk = mask[kcol] != 0;
      #pragma unroll
      for (int r = 0; r < 8; ++r) {
        float v = s[nt][r] + lds_pair[wave][buf][r + (hi << 3)][16 * nt + lo];
        s[nt][r] = mk ? v : -1.0e9f;
      }
    }

    // online softmax across this 32-key slab
    #pragma unroll
    for (int r = 0; r < 8; ++r) {
      float mt = fmaxf(s[0][r], s[1][r]);
      #pragma unroll
      for (int off = 8; off >= 1; off >>= 1)
        mt = fmaxf(mt, __shfl_xor(mt, off, 32));
      const float mn    = fmaxf(m_r[r], mt);
      const float alpha = __expf(m_r[r] - mn);
      m_r[r] = mn;
      const float p0 = __expf(s[0][r] - mn);
      const float p1 = __expf(s[1][r] - mn);
      s[0][r] = p0; s[1][r] = p1;
      float rs = p0 + p1;
      #pragma unroll
      for (int off = 8; off >= 1; off >>= 1)
        rs += __shfl_xor(rs, off, 32);
      l_r[r] = l_r[r] * alpha + rs;
      o[0][r] *= alpha; o[1][r] *= alpha; o[2][r] *= alpha;
    }

    // C-layout P -> LDS -> A-layout bf16 fragment (cross-lane re-layout)
    #pragma unroll
    for (int nt = 0; nt < 2; ++nt)
      #pragma unroll
      for (int r = 0; r < 8; ++r)
        lds_p[wave][r + (hi << 3)][16 * nt + lo] = s[nt][r];
    asm volatile("s_wait_dscnt 0x0" ::: "memory");
    bf16x16 pa;
    #pragma unroll
    for (int e = 0; e < 8; ++e) {
      pa[e]     = (__bf16)lds_p[wave][lo][(hi << 4) + e];
      pa[8 + e] = (__bf16)lds_p[wave][lo][(hi << 4) + 8 + e];
    }

    // O += P V  (only the 3 real d-tiles; tile 3 is padding)
    #pragma unroll
    for (int j = 0; j < 3; ++j) {
      const __bf16* vp =
          vtbuf + (size_t)(head * D_PAD + 16 * j + lo) * N_SEQ + k0 + (hi << 4);
      bf16x16 vb = *(const bf16x16*)vp;  // contiguous 32B
      o[j] = wmma_bf16(pa, vb, o[j]);
    }
  }

  // normalize, gate, store (b=1, layout [n][h*d])
  #pragma unroll
  for (int j = 0; j < 3; ++j) {
    #pragma unroll
    for (int r = 0; r < 8; ++r) {
      const int row = q0 + r + (hi << 3);
      const int col = head * D_HEAD + 16 * j + lo;
      const float val = o[j][r] / l_r[r];
      out[(size_t)row * C_DIM + col] = val * gate[(size_t)row * C_DIM + col];
    }
  }
}

// ---------------------------------------------------------------------------
extern "C" void kernel_launch(void* const* d_in, const int* in_sizes, int n_in,
                              void* d_out, int out_size, void* d_ws, size_t ws_size,
                              hipStream_t stream) {
  const float*         x1   = (const float*)d_in[0];
  const float*         x2   = (const float*)d_in[1];
  const unsigned char* mask = (const unsigned char*)d_in[2];
  const float*         pair = (const float*)d_in[3];
  const float*         Wq   = (const float*)d_in[4];
  const float*         bq   = (const float*)d_in[5];
  const float*         Wk   = (const float*)d_in[6];
  const float*         Wv   = (const float*)d_in[7];
  const float*         Wg   = (const float*)d_in[8];
  float* out = (float*)d_out;

  char* ws = (char*)d_ws;
  const size_t qkv_bytes = (size_t)N_HEAD * N_SEQ * D_PAD * sizeof(__bf16);  // 4 MB
  const size_t gate_bytes = (size_t)N_SEQ * C_DIM * sizeof(float);           // 6 MB
  const size_t x_bytes = (size_t)N_SEQ * C_DIM * sizeof(__bf16);             // 3 MB
  const size_t w_bytes = (size_t)C_DIM * C_DIM * sizeof(__bf16);             // 1.1 MB
  __bf16* qbuf  = (__bf16*)(ws);
  __bf16* kbuf  = (__bf16*)(ws + qkv_bytes);
  __bf16* vtbuf = (__bf16*)(ws + 2 * qkv_bytes);
  float*  gate  = (float*)(ws + 3 * qkv_bytes);
  __bf16* xb1   = (__bf16*)(ws + 3 * qkv_bytes + gate_bytes);
  __bf16* xb2   = (__bf16*)(ws + 3 * qkv_bytes + gate_bytes + x_bytes);
  __bf16* wb    = (__bf16*)(ws + 3 * qkv_bytes + gate_bytes + 2 * x_bytes);

  const size_t n16 = (3 * qkv_bytes) / 16;
  zero_ws<<<dim3((unsigned)((n16 + 255) / 256)), 256, 0, stream>>>((uint4*)ws, n16);

  const int xn4 = (N_SEQ * C_DIM) / 4, wn4 = (C_DIM * C_DIM) / 4;
  cvt_bf16<<<(xn4 + 255) / 256, 256, 0, stream>>>(x1, xb1, xn4);
  cvt_bf16<<<(xn4 + 255) / 256, 256, 0, stream>>>(x2, xb2, xn4);
  cvt_bf16<<<(wn4 + 255) / 256, 256, 0, stream>>>(Wq, wb + 0 * (size_t)C_DIM * C_DIM, wn4);
  cvt_bf16<<<(wn4 + 255) / 256, 256, 0, stream>>>(Wk, wb + 1 * (size_t)C_DIM * C_DIM, wn4);
  cvt_bf16<<<(wn4 + 255) / 256, 256, 0, stream>>>(Wv, wb + 2 * (size_t)C_DIM * C_DIM, wn4);
  cvt_bf16<<<(wn4 + 255) / 256, 256, 0, stream>>>(Wg, wb + 3 * (size_t)C_DIM * C_DIM, wn4);

  proj_kernel<<<dim3(N_SEQ / 128, C_DIM / 64, 4), 256, 0, stream>>>(
      xb1, xb2, wb, bq, qbuf, kbuf, vtbuf, gate);

  attn_kernel<<<dim3(N_SEQ / 128, N_HEAD), 256, 0, stream>>>(
      qbuf, kbuf, vtbuf, gate, pair, mask, out);
}
